// xLSTM_74208444940491
// MI455X (gfx1250) — matrix operand order repaired and checked by reading, not compiled, not run on hardware
//
#include <hip/hip_runtime.h>
#include <hip/hip_bf16.h>

// ---------------------------------------------------------------------------
// CDNA5 (gfx1250, wave32) xLSTM: bf16 WMMA GEMMs + persistent LDS-resident
// recurrent scan + fused GELU/residual/LayerNorm epilogue.
// ---------------------------------------------------------------------------

typedef __attribute__((ext_vector_type(16))) __bf16      v16bf;
typedef __attribute__((ext_vector_type(8)))  float       v8f;
typedef __attribute__((ext_vector_type(4)))  unsigned    v4u;

// ---- WMMA fragment loaders (bf16, row-major sources, 16B-aligned) ----------
// A fragment: 16x32 (MxK). Lane m=lane&15, half=lane>>4.
// Each lane takes K = half*16 + {0..15} (matches B fragment)
// -> 32 contiguous bytes.
__device__ __forceinline__ v16bf load_a_frag(const __bf16* __restrict__ src, int ld) {
    int lane = threadIdx.x & 31;
    int m = lane & 15;
    int half = lane >> 4;
    const __bf16* p0 = src + (long)m * ld + half * 16;
    union { v16bf v; v4u q[2]; } u;
    u.q[0] = *(const v4u*)(p0);
    u.q[1] = *(const v4u*)(p0 + 8);
    return u.v;
}

// B fragment: 32x16 (KxN) from B^T storage (N rows x K cols, row-major).
// Lane n=lane&15 covers column n; lanes>=16 take K 16..31: 32 contiguous bytes.
__device__ __forceinline__ v16bf load_b_frag(const __bf16* __restrict__ srcNK, int ld) {
    int lane = threadIdx.x & 31;
    int n = lane & 15;
    int kb = (lane >> 4) * 16;
    const v4u* p = (const v4u*)(srcNK + (long)n * ld + kb);
    union { v16bf v; v4u q[2]; } u;
    u.q[0] = p[0];
    u.q[1] = p[1];
    return u.v;
}

// ---- f32 -> bf16 conversion ------------------------------------------------
__global__ void cvt_f32_bf16(const float* __restrict__ in, __bf16* __restrict__ out, long n) {
    long i = (long)blockIdx.x * blockDim.x + threadIdx.x;
    if (i < n) out[i] = (__bf16)in[i];
}

// ---- Generic GEMM: C(MxN,f32) = A(MxK,bf16) * B(N,K bf16)^T + bias ---------
// 256 threads = 8 waves; each wave computes a 16x64 strip (4 accumulators),
// reusing its A fragment across 4 WMMAs per K-step.
__global__ void __launch_bounds__(256) gemm_bf16(
    const __bf16* __restrict__ A, const __bf16* __restrict__ Bt,
    const float* __restrict__ bias, float* __restrict__ C,
    int M, int N, int K)
{
    int wave = blockIdx.x * (blockDim.x >> 5) + (threadIdx.x >> 5);
    int ntiles = N / 64;
    int mt = wave / ntiles;
    int nt = wave % ntiles;
    if (mt >= M / 16) return;          // wave-uniform
    int m0 = mt * 16, n0 = nt * 64;

    v8f acc[4] = {};
    const __bf16* Arow = A + (long)m0 * K;
    for (int k0 = 0; k0 < K; k0 += 32) {
        if (k0 + 32 < K) {
            __builtin_prefetch(Arow + k0 + 32, 0, 1);
            __builtin_prefetch(Bt + (long)n0 * K + k0 + 32, 0, 1);
        }
        v16bf a = load_a_frag(Arow + k0, K);
#pragma unroll
        for (int j = 0; j < 4; ++j) {
            v16bf b = load_b_frag(Bt + (long)(n0 + j * 16) * K + k0, K);
            acc[j] = __builtin_amdgcn_wmma_f32_16x16x32_bf16(
                false, a, false, b, (short)0, acc[j], false, false);
        }
    }
    // C/D layout: VGPR r -> (M=r, N=lane) lanes0-15 ; (M=8+r, N=lane-16) lanes16-31
    int lane = threadIdx.x & 31;
    int n = lane & 15;
    int mo = (lane >> 4) * 8;
#pragma unroll
    for (int j = 0; j < 4; ++j) {
        float bv = bias[n0 + j * 16 + n];
#pragma unroll
        for (int r = 0; r < 8; ++r) {
            C[(long)(m0 + mo + r) * N + n0 + j * 16 + n] = acc[j][r] + bv;
        }
    }
}

// ---- scan init -------------------------------------------------------------
__global__ void init_scan(__bf16* hbuf, unsigned int* syncc) {
    int i = blockIdx.x * blockDim.x + threadIdx.x;
    if (i < 16 * 1024) hbuf[i] = (__bf16)0.f;
    if (i == 0) *syncc = 0u;
}

// ---- Persistent sLSTM scan -------------------------------------------------
// 64 WGs x 128 threads (4 waves). WG jt owns hidden columns j0..j0+15.
// Wave g handles gate g (N offset g*H). R slice (4 x 16 x 1024 bf16 = 128KB)
// lives in LDS for the whole scan; h (16x1024 bf16, rows 8..15 zero) lives in
// global and is grid-synced each timestep via an L2 atomic counter.
#define SCAN_LDS_BYTES (4 * 16 * 1024 * 2 + 4 * 8 * 16 * 4)

__global__ void __launch_bounds__(128) slstm_scan(
    const __bf16* __restrict__ Rbf,   // (4H, H) row-major bf16
    const float*  __restrict__ xp,    // (B*S, 4H) f32, row = b*S + s
    __bf16*       __restrict__ hseq,  // (B*S, H) bf16 out
    __bf16*       __restrict__ hbuf,  // (16, H) bf16, rows 8..15 zero
    unsigned int* __restrict__ syncc,
    int S)
{
    const int H = 1024, G = 4096;
    extern __shared__ char smem[];
    __bf16* Rs      = (__bf16*)smem;                       // 4*16*1024 bf16
    float*  pre_lds = (float*)(smem + 4 * 16 * 1024 * 2);  // [gate][8][16]

    int jt   = blockIdx.x;        // 0..63
    int j0   = jt * 16;
    int wave = threadIdx.x >> 5;  // gate index 0..3
    int lane = threadIdx.x & 31;

    // Preload this WG's R slice: gate 'wave', rows (wave*H + j0 .. +15), 1024 K.
    {
        const v4u* src = (const v4u*)(Rbf + (long)(wave * H + j0) * H);
        v4u* dst = (v4u*)(Rs + wave * 16 * 1024);
        for (int i = lane; i < 16 * 1024 / 8; i += 32) dst[i] = src[i];
    }
    __syncthreads();

    // per-thread recurrent state for element (b, j0+j)
    int b = threadIdx.x >> 4;     // 0..7
    int j = threadIdx.x & 15;     // 0..15
    float c = 0.f, nrm = 0.f, mst = 0.f;
    unsigned int nblocks = gridDim.x;

    const __bf16* Rw = Rs + wave * 16 * 1024;

    for (int t = 0; t < S; ++t) {
        // pre[gate] tile = h(16x1024,bf16) x R_slice^T, K=1024
        v8f acc = {};
        for (int k0 = 0; k0 < 1024; k0 += 32) {
            v16bf a = load_a_frag(hbuf + k0, 1024);   // global (rows 8..15 zero)
            v16bf bb = load_b_frag(Rw + k0, 1024);    // LDS -> ds_load_b128
            acc = __builtin_amdgcn_wmma_f32_16x16x32_bf16(
                false, a, false, bb, (short)0, acc, false, false);
        }
        // add xp and publish the 8x16 valid tile (lanes 0-15 hold M=0..7)
        if (lane < 16) {
#pragma unroll
            for (int r = 0; r < 8; ++r) {
                float p = acc[r] + xp[((long)r * S + t) * G + wave * H + j0 + lane];
                pre_lds[(wave * 8 + r) * 16 + lane] = p;
            }
        }
        __syncthreads();

        // exponential-gated, log-stabilized update
        float i_pre = pre_lds[(0 * 8 + b) * 16 + j];
        float f_pre = pre_lds[(1 * 8 + b) * 16 + j];
        float z_pre = pre_lds[(2 * 8 + b) * 16 + j];
        float o_pre = pre_lds[(3 * 8 + b) * 16 + j];
        float m_new = fmaxf(f_pre + mst, i_pre);
        float ig = __expf(i_pre - m_new);
        float fg = __expf(f_pre + mst - m_new);
        float cn = fg * c + ig * tanhf(z_pre);
        float nn = fg * nrm + ig;
        float hv = (1.f / (1.f + __expf(-o_pre))) * (cn / nn);
        c = cn; nrm = nn; mst = m_new;

        __bf16 hb = (__bf16)hv;
        hbuf[b * H + j0 + j] = hb;                   // for next step (all WGs)
        hseq[((long)b * S + t) * H + j0 + j] = hb;   // sequence output
        __syncthreads();

        // grid-wide barrier (release h, acquire others' h)
        __threadfence();
        __syncthreads();
        if (threadIdx.x == 0) {
            __hip_atomic_fetch_add(syncc, 1u, __ATOMIC_ACQ_REL, __HIP_MEMORY_SCOPE_AGENT);
            unsigned int target = nblocks * (unsigned int)(t + 1);
            while (__hip_atomic_load(syncc, __ATOMIC_ACQUIRE, __HIP_MEMORY_SCOPE_AGENT) < target) {
                __builtin_amdgcn_s_sleep(2);
            }
        }
        __syncthreads();
        // acquire fence at agent scope: invalidates L0 so the next step's
        // WMMA A-loads observe every WG's freshly written h.
        __builtin_amdgcn_fence(__ATOMIC_ACQUIRE, "agent");
    }
}

// ---- GELU(exact) + residual + LayerNorm ------------------------------------
__global__ void __launch_bounds__(256) gelu_res_ln(
    const float* __restrict__ y, const float* __restrict__ xres,
    const float* __restrict__ gamma, const float* __restrict__ beta,
    float* __restrict__ xout, int D)
{
    long row = blockIdx.x;
    const float* yr = y + row * D;
    const float* xr = xres + row * D;
    __shared__ float red0[8], red1[8];

    float v[4];
    float s = 0.f, s2 = 0.f;
#pragma unroll
    for (int e = 0; e < 4; ++e) {
        int idx = threadIdx.x + e * 256;
        float t = yr[idx];
        float gl = 0.5f * t * (1.f + erff(t * 0.70710678118654752f));
        float r = gl + xr[idx];
        v[e] = r; s += r; s2 += r * r;
    }
    int lane = threadIdx.x & 31, wv = threadIdx.x >> 5;
#pragma unroll
    for (int o = 16; o > 0; o >>= 1) { s += __shfl_down(s, o); s2 += __shfl_down(s2, o); }
    if (lane == 0) { red0[wv] = s; red1[wv] = s2; }
    __syncthreads();
    if (threadIdx.x == 0) {
        float a = 0.f, b2 = 0.f;
        for (int i = 0; i < 8; ++i) { a += red0[i]; b2 += red1[i]; }
        float mu = a / D;
        float var = b2 / D - mu * mu;
        red0[0] = mu;
        red1[0] = rsqrtf(var + 1e-5f);
    }
    __syncthreads();
    float mu = red0[0], rs = red1[0];
#pragma unroll
    for (int e = 0; e < 4; ++e) {
        int idx = threadIdx.x + e * 256;
        xout[row * D + idx] = (v[e] - mu) * rs * gamma[idx] + beta[idx];
    }
}

// ---------------------------------------------------------------------------
extern "C" void kernel_launch(void* const* d_in, const int* in_sizes, int n_in,
                              void* d_out, int out_size, void* d_ws, size_t ws_size,
                              hipStream_t stream) {
    (void)in_sizes; (void)n_in; (void)out_size; (void)ws_size;
    const int B = 8, S = 1024, D = 1024, H = 1024, G = 4096;
    const long MS = (long)B * S; // 8192 rows

    char* ws = (char*)d_ws;
    size_t off = 0;
    auto alloc = [&](size_t bytes) -> void* {
        void* p = ws + off;
        off += (bytes + 255) & ~(size_t)255;
        return p;
    };
    float*   xp    = (float*)  alloc((size_t)MS * G * 4);   // 128 MB
    __bf16*  hseq  = (__bf16*) alloc((size_t)MS * H * 2);   // 16 MB
    __bf16*  xbf   = (__bf16*) alloc((size_t)MS * D * 2);   // 16 MB
    __bf16*  Wbf   = (__bf16*) alloc((size_t)G * D * 2);    // 8 MB
    __bf16*  Rbf   = (__bf16*) alloc((size_t)G * H * 2);    // 8 MB
    __bf16*  pWbf  = (__bf16*) alloc((size_t)D * H * 2);    // 2 MB
    float*   ybuf  = (float*)  alloc((size_t)MS * H * 4);   // 32 MB
    float*   xmid  = (float*)  alloc((size_t)MS * D * 4);   // 32 MB
    __bf16*  hbuf  = (__bf16*) alloc(16 * 1024 * 2);
    unsigned int* syncc = (unsigned int*)alloc(256);

    const float* input_seq = (const float*)d_in[0];

    // persistent scan needs 130 KB dynamic LDS
    (void)hipFuncSetAttribute((const void*)slstm_scan,
                              hipFuncAttributeMaxDynamicSharedMemorySize,
                              SCAN_LDS_BYTES);

    for (int blk = 0; blk < 2; ++blk) {
        const float* Wf  = (const float*)d_in[1 + 7 * blk + 0];
        const float* Rf  = (const float*)d_in[1 + 7 * blk + 1];
        const float* bf  = (const float*)d_in[1 + 7 * blk + 2];
        const float* pWf = (const float*)d_in[1 + 7 * blk + 3];
        const float* pbf = (const float*)d_in[1 + 7 * blk + 4];
        const float* gmf = (const float*)d_in[1 + 7 * blk + 5];
        const float* btf = (const float*)d_in[1 + 7 * blk + 6];
        const float* xin  = (blk == 0) ? input_seq : xmid;
        float*       xout = (blk == 0) ? xmid : (float*)d_out;

        long nx = MS * (long)D;
        cvt_f32_bf16<<<(int)((nx + 255) / 256), 256, 0, stream>>>(xin, xbf, nx);
        long nw = (long)G * D;
        cvt_f32_bf16<<<(int)((nw + 255) / 256), 256, 0, stream>>>(Wf, Wbf, nw);
        cvt_f32_bf16<<<(int)((nw + 255) / 256), 256, 0, stream>>>(Rf, Rbf, nw);
        long npw = (long)D * H;
        cvt_f32_bf16<<<(int)((npw + 255) / 256), 256, 0, stream>>>(pWf, pWbf, npw);

        // xp = x * W^T + b   (M=8192, N=4096, K=1024)
        {
            int waves = (int)(MS / 16) * (G / 64);
            gemm_bf16<<<waves / 8, 256, 0, stream>>>(xbf, Wbf, bf, xp, (int)MS, G, D);
        }

        // sequential exponential-gated scan, R resident in LDS
        init_scan<<<(16 * 1024 + 255) / 256, 256, 0, stream>>>(hbuf, syncc);
        slstm_scan<<<64, 128, SCAN_LDS_BYTES, stream>>>(Rbf, xp, hseq, hbuf, syncc, S);

        // y = hseq * pW^T + pb   (M=8192, N=1024, K=1024)
        {
            int waves = (int)(MS / 16) * (H / 64);
            gemm_bf16<<<waves / 8, 256, 0, stream>>>(hseq, pWbf, pbf, ybuf, (int)MS, H, H);
        }

        // x_next = LN(GELU(y) + x)
        gelu_res_ln<<<(int)MS, 256, 0, stream>>>(ybuf, xin, gmf, btf, xout, D);
    }
}
